// GuidanceController_61787399520373
// MI455X (gfx1250) — matrix operand order repaired and checked by reading, not compiled, not run on hardware
//
#include <hip/hip_runtime.h>
#include <hip/hip_bf16.h>

typedef __attribute__((ext_vector_type(16))) _Float16 v16h;
typedef __attribute__((ext_vector_type(8)))  _Float16 v8h;
typedef __attribute__((ext_vector_type(8)))  float    v8f;

#define GRID_RES 128
#define HALO     8
#define ROWS     (GRID_RES + 2 * HALO)   // 144
#define PITCH    152                     // halves/row: 304 B, 16B-aligned, bank-friendly
#define BUFSZ    (ROWS * PITCH)
#define ITERS    50
#define TILE     16
#define NTILES   8

// ---------------------------------------------------------------------------
// One workgroup (256 threads = 8 wave32) per field n. 50 Jacobi iterations of
// the screened Poisson solve run entirely in LDS. Per 16x16 tile:
//   acc  = U(16x32 row window)  x Sband(32x16 const)   [horizontal neighbors]
//   acc += Sband(16x32 const)   x U(32x16 col window)  [vertical neighbors]
// via two v_wmma_f32_16x16x32_f16. Band fragments have 1.0 at K = idx+7,idx+9
// (window offset -8), so out-of-grid taps hit the 8-cell zero halo.
// Field kept in BOTH row-major (U) and col-major (UT) LDS copies so every
// operand is a contiguous, 16B-aligned run -> ds_load_b128 / ds_store_b128.
// The D fragment is a per-lane column, i.e. contiguous in UT: the transposed
// copy costs one b128 store per tile.
// ---------------------------------------------------------------------------
__global__ __launch_bounds__(256) void guidance_field_kernel(
    const float* __restrict__ pos,    // (N, T, 2)
    const float* __restrict__ gw,     // (N, 3)
    const float* __restrict__ head,   // (N, 2)
    float* __restrict__ out_per_n,    // (N)
    int T)
{
  // [0]=U row-major, [1]=U col-major, [2]/[3]=ping-pong partners
  __shared__ __align__(16) _Float16 smem[4][BUFSZ];
  __shared__ float sred[256];

  const int n    = blockIdx.x;
  const int tid  = threadIdx.x;
  const int lane = tid & 31;
  const int wave = tid >> 5;          // tile-row owned by this wave
  const int hl   = (lane < 16) ? 0 : 1;
  const int m16  = lane & 15;

  const float DX     = 100.0f / 127.0f;
  const float invden = 1.0f / (4.0f + 0.1f * 0.1f * DX * DX);

  // L2 prefetch of this block's trajectory slice; consumed only after the
  // 50-iteration LDS solve, so HBM latency is fully hidden.
  __builtin_prefetch(pos + (size_t)n * T * 2, 0, 1);

  // --- goal cell (redundant per-thread; broadcast loads) -------------------
  const float p0x = pos[(size_t)n * T * 2 + 0];
  const float p0y = pos[(size_t)n * T * 2 + 1];
  const float hx = head[n * 2 + 0], hy = head[n * 2 + 1];
  const float hn = fmaxf(sqrtf(hx * hx + hy * hy), 1e-12f);
  const float hnx = hx / hn, hny = hy / hn;
  const float gx = (fminf(fmaxf(p0x + hnx * 50.0f, -50.0f), 50.0f) + 50.0f) / DX;
  const float gy = (fminf(fmaxf(p0y + hny * 50.0f, -50.0f), 50.0f) + 50.0f) / DX;
  const int gxi = (int)gx;
  const int gyi = (int)gy;
  const int goal_rm = (gyi + HALO) * PITCH + (gxi + HALO);  // row-major idx
  const int goal_cm = (gxi + HALO) * PITCH + (gyi + HALO);  // col-major idx

  // --- init all buffers: zero halo, 0.5 interior ---------------------------
  for (int i = tid; i < BUFSZ; i += 256) {
    const int r = i / PITCH, c = i % PITCH;
    const bool interior = (r >= HALO) & (r < HALO + GRID_RES) &
                          (c >= HALO) & (c < HALO + GRID_RES);
    const _Float16 v = interior ? (_Float16)0.5f : (_Float16)0.0f;
    smem[0][i] = v; smem[1][i] = v; smem[2][i] = v; smem[3][i] = v;
  }
  __syncthreads();
  if (tid == 0) { smem[0][goal_rm] = (_Float16)1.0f; smem[1][goal_cm] = (_Float16)1.0f; }
  __syncthreads();

  // --- constant band fragments (exact 0/1 in f16) --------------------------
  // A-layout (16x32): elem i -> K = i + 8*hl + (i<8 ? 0 : 8), row m16.
  // B-layout (32x16): elem i -> K = i + 8*hl + (i<8 ? 0 : 8), col m16.
  v16h Sa, Sb;
#pragma unroll
  for (int i = 0; i < 16; ++i) {
    const int Ka = i + hl * 8 + ((i < 8) ? 0 : 8);
    Sa[i] = (Ka == m16 + 7 || Ka == m16 + 9) ? (_Float16)1.0f : (_Float16)0.0f;
    const int Kb = i + hl * 8 + ((i < 8) ? 0 : 8);
    Sb[i] = (Kb == m16 + 7 || Kb == m16 + 9) ? (_Float16)1.0f : (_Float16)0.0f;
  }

  // --- 50 Jacobi iterations on the WMMA pipe -------------------------------
  int src = 0;
  for (int it = 0; it < ITERS; ++it) {
    const _Float16* __restrict__ U   = smem[src * 2 + 0];
    const _Float16* __restrict__ UT  = smem[src * 2 + 1];
    _Float16* __restrict__       Un  = smem[(src ^ 1) * 2 + 0];
    _Float16* __restrict__       UnT = smem[(src ^ 1) * 2 + 1];
    const int I = wave;
#pragma unroll 2
    for (int J = 0; J < NTILES; ++J) {
      // A fragment: rows 16I+m, col window [16J-8,16J+24) -> two b128 loads
      const _Float16* rowp =
          U + (TILE * I + HALO + m16) * PITCH + TILE * J + hl * 8;
      const v8h alo = *(const v8h*)(rowp);
      const v8h ahi = *(const v8h*)(rowp + 16);
      const v16h Au = __builtin_shufflevector(alo, ahi,
          0, 1, 2, 3, 4, 5, 6, 7, 8, 9, 10, 11, 12, 13, 14, 15);
      // B fragment: row window [16I-8,16I+24), col 16J+m: contiguous in UT
      const _Float16* colp =
          UT + (TILE * J + HALO + m16) * PITCH + TILE * I + hl * 8;
      const v8h blo = *(const v8h*)(colp);
      const v8h bhi = *(const v8h*)(colp + 16);
      const v16h Bu = __builtin_shufflevector(blo, bhi,
          0, 1, 2, 3, 4, 5, 6, 7, 8, 9, 10, 11, 12, 13, 14, 15);

      v8f acc = {};
      acc = __builtin_amdgcn_wmma_f32_16x16x32_f16(false, Au, false, Sb,
                                                   (short)0, acc, false, false);
      acc = __builtin_amdgcn_wmma_f32_16x16x32_f16(false, Sa, false, Bu,
                                                   (short)0, acc, false, false);

      // D: elem r -> (tile row r + 8*hl, tile col m16). Scale, pack to f16.
      v8h d;
#pragma unroll
      for (int r = 0; r < 8; ++r) d[r] = (_Float16)(acc[r] * invden);
      // transposed copy: per-lane column is contiguous -> one b128 store
      *(v8h*)(UnT + (TILE * J + HALO + m16) * PITCH + TILE * I + HALO + hl * 8) = d;
      // row-major copy: 8 strided b16 stores
#pragma unroll
      for (int r = 0; r < 8; ++r) {
        Un[(TILE * I + HALO + hl * 8 + r) * PITCH + TILE * J + HALO + m16] = d[r];
      }
    }
    __syncthreads();
    if (tid == 0) {           // Dirichlet goal cell, both orientations
      Un[goal_rm]  = (_Float16)1.0f;
      UnT[goal_cm] = (_Float16)1.0f;
    }
    __syncthreads();
    src ^= 1;
  }

  // --- bilinear sampling of the converged field ----------------------------
  const _Float16* __restrict__ F = smem[src * 2 + 0];
  float lsum = 0.0f;
  for (int t = tid; t < T; t += 256) {
    const float px = pos[((size_t)n * T + t) * 2 + 0];
    const float py = pos[((size_t)n * T + t) * 2 + 1];
    float x = fminf(fmaxf((px + 50.0f) * (127.0f / 100.0f), 0.0f), 127.0f);
    float y = fminf(fmaxf((py + 50.0f) * (127.0f / 100.0f), 0.0f), 127.0f);
    const float x0 = floorf(x), y0 = floorf(y);
    const float wx = x - x0, wy = y - y0;
    int x0i = (int)x0; x0i = (x0i < 0) ? 0 : ((x0i > 127) ? 127 : x0i);
    int y0i = (int)y0; y0i = (y0i < 0) ? 0 : ((y0i > 127) ? 127 : y0i);
    const int x1i = (x0i + 1 > 127) ? 127 : x0i + 1;
    const int y1i = (y0i + 1 > 127) ? 127 : y0i + 1;
    const float v00 = (float)F[(y0i + HALO) * PITCH + x0i + HALO];
    const float v01 = (float)F[(y0i + HALO) * PITCH + x1i + HALO];
    const float v10 = (float)F[(y1i + HALO) * PITCH + x0i + HALO];
    const float v11 = (float)F[(y1i + HALO) * PITCH + x1i + HALO];
    lsum += v00 * (1.0f - wx) * (1.0f - wy) + v01 * wx * (1.0f - wy) +
            v10 * (1.0f - wx) * wy + v11 * wx * wy;
  }
  sred[tid] = lsum;
  __syncthreads();
  for (int s = 128; s > 0; s >>= 1) {
    if (tid < s) sred[tid] += sred[tid + s];
    __syncthreads();
  }

  if (tid == 0) {
    const float w_lane = gw[n * 3 + 0];
    const float w_left = gw[n * 3 + 1];
    const float w_rght = gw[n * 3 + 2];
    const float pTx = pos[((size_t)n * T + (T - 1)) * 2 + 0];
    const float pTy = pos[((size_t)n * T + (T - 1)) * 2 + 1];
    const float dx = pTx - p0x, dy = pTy - p0y;
    const float ld = dx * (-hny) + dy * hnx;            // left drift
    const float e_left = -ld + 2.0f * fmaxf(-ld, 0.0f);
    const float e_rght =  ld + 2.0f * fmaxf( ld, 0.0f);
    const float e_lane = (w_lane > 0.0001f) ? -sred[0] : 0.0f;
    out_per_n[n] = w_lane * e_lane + w_left * e_left + w_rght * e_rght;
  }
}

// Deterministic tree reduction (no float atomics: bit-identical replays).
__global__ __launch_bounds__(256) void reduce_kernel(
    const float* __restrict__ in, float* __restrict__ out, int N)
{
  __shared__ float s[256];
  float v = 0.0f;
  for (int i = threadIdx.x; i < N; i += 256) v += in[i];
  s[threadIdx.x] = v;
  __syncthreads();
  for (int st = 128; st > 0; st >>= 1) {
    if (threadIdx.x < st) s[threadIdx.x] += s[threadIdx.x + st];
    __syncthreads();
  }
  if (threadIdx.x == 0) out[0] = s[0];
}

extern "C" void kernel_launch(void* const* d_in, const int* in_sizes, int n_in,
                              void* d_out, int out_size, void* d_ws, size_t ws_size,
                              hipStream_t stream) {
  const float* pos  = (const float*)d_in[0];  // agent_positions_flat (N,T,2)
  const float* gw   = (const float*)d_in[1];  // guidance_weights    (N,3)
  const float* head = (const float*)d_in[2];  // initial_headings    (N,2)
  // d_in[3] map_points is unused by the reference.

  const int N = in_sizes[1] / 3;              // 384
  const int T = in_sizes[0] / (2 * N);        // 80

  float* per_n = (float*)d_ws;                // N floats of scratch
  guidance_field_kernel<<<N, 256, 0, stream>>>(pos, gw, head, per_n, T);
  reduce_kernel<<<1, 256, 0, stream>>>(per_n, (float*)d_out, N);
}
